// Attention_27994596835718
// MI455X (gfx1250) — matrix operand order repaired and checked by reading, not compiled, not run on hardware
//
#include <hip/hip_runtime.h>
#include <hip/hip_bf16.h>

#define Bb 16
#define Cc 512
#define Npix 4096   // 64*64

typedef __attribute__((ext_vector_type(16))) __bf16 v16bf;
typedef __attribute__((ext_vector_type(8)))  __bf16 v8bf;
typedef __attribute__((ext_vector_type(8)))  float  v8f;
typedef __attribute__((ext_vector_type(8)))  float  f32x8;

// ---- CDNA5 async global->LDS copy (ASYNCcnt) --------------------------------

__device__ __forceinline__ void async_copy_b128(const void* src, void* dst_lds) {
  // VDST = per-lane LDS byte address (low 32 bits of generic shared pointer),
  // VADDR = 64-bit global address.  Tracked by ASYNCcnt.
  unsigned lds = (unsigned)(unsigned long long)dst_lds;
  asm volatile("global_load_async_to_lds_b128 %0, %1, off"
               :: "v"(lds), "v"(src) : "memory");
}

__device__ __forceinline__ void wait_async0() {
  asm volatile("s_wait_asynccnt 0x0" ::: "memory");
}

// Stage a 16-row x 32-col bf16 tile (1KB) into LDS: 64 b128 chunks, threads 0..63.
__device__ __forceinline__ void stage_tile(const __bf16* __restrict__ src_row0,
                                           size_t ld, __bf16* dst, int tid) {
  if (tid < 64) {
    const int r = tid >> 2, ch = tid & 3;
    async_copy_b128(src_row0 + (size_t)r * ld + ch * 8, dst + tid * 8);
  }
}

// ---- WMMA helpers (layouts per CDNA5 ISA 7.12.2) -----------------

__device__ __forceinline__ v8f wmma_bf16(v16bf a, v16bf b, v8f c) {
  return __builtin_amdgcn_wmma_f32_16x16x32_bf16(false, a, false, b, (short)0, c,
                                                 false, false);
}

__device__ __forceinline__ v16bf cat8(v8bf lo, v8bf hi) {
  return __builtin_shufflevector(lo, hi, 0, 1, 2, 3, 4, 5, 6, 7,
                                 8, 9, 10, 11, 12, 13, 14, 15);
}

// A fragment from LDS tile [16 rows][32 K] (row-major): lane row = l,
// elements 0..15 -> K = half*16 + i.  Two ds_load_b128.
__device__ __forceinline__ v16bf load_a_frag_lds(const __bf16* tile, int l, int half) {
  const __bf16* p = tile + l * 32;
  const v8bf lo = *(const v8bf*)(p + half * 16);
  const v8bf hi = *(const v8bf*)(p + half * 16 + 8);
  return cat8(lo, hi);
}

// B fragment from a K-major matrix: p = colbase + k0 + half*16.
// Element i -> K = half*16 + i : one contiguous 32B region -> 2 b128 loads.
__device__ __forceinline__ v16bf load_b_frag_kmajor(const __bf16* __restrict__ p) {
  const v8bf lo = *(const v8bf*)p;
  const v8bf hi = *(const v8bf*)(p + 8);
  return cat8(lo, hi);
}

// ---- Kernel 0a: x f32 -> bf16 (streamed, nontemporal reads) -----------------

__global__ __launch_bounds__(256) void xconv_kernel(const float* __restrict__ x,
                                                    __bf16* __restrict__ xb) {
  const size_t i = ((size_t)blockIdx.x * 256 + threadIdx.x) * 8;
  const f32x8 f = __builtin_nontemporal_load((const f32x8*)(x + i));
  v8bf o;
#pragma unroll
  for (int j = 0; j < 8; ++j) o[j] = (__bf16)f[j];
  *(v8bf*)(xb + i) = o;
}

// ---- Kernel 0b: weights f32 -> bf16, stored K-major (transposed) ------------

__global__ __launch_bounds__(256) void wconv_kernel(const float* __restrict__ wq,
                                                    const float* __restrict__ wk,
                                                    const float* __restrict__ wv,
                                                    __bf16* __restrict__ wbfT) {
  const int i = blockIdx.x * 256 + threadIdx.x;  // 0 .. 3*C*C
  const int sz = Cc * Cc;
  const int mat = (i < sz) ? 0 : (i < 2 * sz) ? 1 : 2;
  const int j = i - mat * sz;                     // j = c*C + f
  const float* src = (mat == 0) ? wq : (mat == 1) ? wk : wv;
  const int c = j >> 9, f = j & (Cc - 1);
  wbfT[(size_t)mat * sz + (size_t)f * Cc + c] = (__bf16)src[j];
}

// ---- Kernel 1: fused QKV GEMM ------------------------------------------------
// Per batch: [N x C] = xb[N x C] * W[C x C] + bias.  q,k stored TRANSPOSED
// ([C x N], K-major for the score GEMM); v stored row-major.

__global__ __launch_bounds__(256) void qkv_gemm_kernel(const __bf16* __restrict__ xb,
                                                       const __bf16* __restrict__ wbfT,
                                                       const float* __restrict__ bq,
                                                       const float* __restrict__ bk,
                                                       const float* __restrict__ bv,
                                                       __bf16* __restrict__ qT,
                                                       __bf16* __restrict__ kT,
                                                       __bf16* __restrict__ vb) {
  __shared__ __align__(16) __bf16 sA[2][16 * 32];

  const int tid  = threadIdx.x;
  const int wave = tid >> 5, lane = tid & 31;
  const int half = lane >> 4, l = lane & 15;

  const int mt  = blockIdx.x;          // 16-pixel row tile
  const int b   = blockIdx.y & 15;
  const int mat = blockIdx.y >> 4;     // 0=q 1=k 2=v
  const int col0 = wave * 64;

  const __bf16* Wt  = wbfT + (size_t)mat * Cc * Cc;   // K-major [f][c]
  const float* bias = (mat == 0) ? bq : (mat == 1) ? bk : bv;

  const __bf16* Arow0 = xb + ((size_t)b * Npix + mt * 16) * Cc;

  v8f z = {};
  v8f acc0 = z, acc1 = z, acc2 = z, acc3 = z;

  stage_tile(Arow0, Cc, sA[0], tid);
  int cur = 0;
  for (int k0 = 0; k0 < Cc; k0 += 32) {
    wait_async0();
    __syncthreads();
    if (k0 + 32 < Cc) stage_tile(Arow0 + k0 + 32, Cc, sA[cur ^ 1], tid);

    const v16bf a = load_a_frag_lds(sA[cur], l, half);
    const __bf16* bp = Wt + (size_t)(col0 + l) * Cc + k0 + half * 16;
    if (k0 + 64 < Cc) __builtin_prefetch(bp + 64, 0, 3);
    acc0 = wmma_bf16(a, load_b_frag_kmajor(bp + (size_t)0 * 16 * Cc), acc0);
    acc1 = wmma_bf16(a, load_b_frag_kmajor(bp + (size_t)1 * 16 * Cc), acc1);
    acc2 = wmma_bf16(a, load_b_frag_kmajor(bp + (size_t)2 * 16 * Cc), acc2);
    acc3 = wmma_bf16(a, load_b_frag_kmajor(bp + (size_t)3 * 16 * Cc), acc3);
    cur ^= 1;
  }

  v8f accs[4] = {acc0, acc1, acc2, acc3};
  if (mat < 2) {
    // transposed store: lane owns column (channel), 8 consecutive pixels -> b128
    __bf16* outT = (mat == 0) ? qT : kT;
#pragma unroll
    for (int t = 0; t < 4; ++t) {
      const int col = col0 + t * 16 + l;
      const float bz = bias[col];
      v8bf pk;
#pragma unroll
      for (int r = 0; r < 8; ++r) pk[r] = (__bf16)(accs[t][r] + bz);
      *(v8bf*)(outT + (size_t)b * Cc * Npix + (size_t)col * Npix +
               mt * 16 + half * 8) = pk;
    }
  } else {
#pragma unroll
    for (int t = 0; t < 4; ++t) {
      const int col = col0 + t * 16 + l;
      const float bz = bias[col];
#pragma unroll
      for (int r = 0; r < 8; ++r) {
        const int row_o = mt * 16 + half * 8 + r;
        vb[((size_t)b * Npix + row_o) * Cc + col] = (__bf16)(accs[t][r] + bz);
      }
    }
  }
}

// ---- Kernel 2: S[b] = Q[b]^T * K[b]  ([C x C], K-dim = Npix) ----------------
// qT/kT are [C x Npix] K-major: both operands load as wide vectors.

__global__ __launch_bounds__(256) void score_gemm_kernel(const __bf16* __restrict__ qT,
                                                         const __bf16* __restrict__ kT,
                                                         float* __restrict__ s) {
  __shared__ __align__(16) __bf16 sA[2][16 * 32];

  const int tid  = threadIdx.x;
  const int wave = tid >> 5, lane = tid & 31;
  const int half = lane >> 4, l = lane & 15;

  const int ct = blockIdx.x;           // 16-channel row tile
  const int b  = blockIdx.y;
  const int col0 = wave * 64;

  const __bf16* Arow0 = qT + (size_t)b * Cc * Npix + (size_t)(ct * 16) * Npix;
  const __bf16* Kb    = kT + (size_t)b * Cc * Npix;

  v8f z = {};
  v8f acc0 = z, acc1 = z, acc2 = z, acc3 = z;

  stage_tile(Arow0, Npix, sA[0], tid);
  int cur = 0;
  for (int n0 = 0; n0 < Npix; n0 += 32) {
    wait_async0();
    __syncthreads();
    if (n0 + 32 < Npix) stage_tile(Arow0 + n0 + 32, Npix, sA[cur ^ 1], tid);

    const v16bf a = load_a_frag_lds(sA[cur], l, half);
    const __bf16* bp = Kb + (size_t)(col0 + l) * Npix + n0 + half * 16;
    if (n0 + 64 < Npix) __builtin_prefetch(bp + 64, 0, 3);
    acc0 = wmma_bf16(a, load_b_frag_kmajor(bp + (size_t)0 * 16 * Npix), acc0);
    acc1 = wmma_bf16(a, load_b_frag_kmajor(bp + (size_t)1 * 16 * Npix), acc1);
    acc2 = wmma_bf16(a, load_b_frag_kmajor(bp + (size_t)2 * 16 * Npix), acc2);
    acc3 = wmma_bf16(a, load_b_frag_kmajor(bp + (size_t)3 * 16 * Npix), acc3);
    cur ^= 1;
  }

  v8f accs[4] = {acc0, acc1, acc2, acc3};
  float* srow = s + (size_t)b * Cc * Cc;
#pragma unroll
  for (int t = 0; t < 4; ++t) {
    const int col = col0 + t * 16 + l;
#pragma unroll
    for (int r = 0; r < 8; ++r)
      srow[(size_t)(ct * 16 + half * 8 + r) * Cc + col] = accs[t][r];
  }
}

// ---- Kernel 3: row softmax + transpose -> aT[d][c] (bf16, K-major) ----------

__global__ __launch_bounds__(256) void softmax_kernel(const float* __restrict__ s,
                                                      __bf16* __restrict__ aT) {
  __shared__ __align__(16) __bf16 ls[8 * Cc];   // 8 rows staged for transpose

  const int wave = threadIdx.x >> 5, lane = threadIdx.x & 31;
  const int grow = blockIdx.x * 8 + wave;       // global row index = b*C + c
  const float* sr = s + (size_t)grow * Cc;

  float vals[16];
  float m = -INFINITY;
#pragma unroll
  for (int j = 0; j < 16; ++j) {
    vals[j] = sr[lane + 32 * j];
    m = fmaxf(m, vals[j]);
  }
#pragma unroll
  for (int off = 16; off > 0; off >>= 1) m = fmaxf(m, __shfl_xor(m, off, 32));

  float sum = 0.f;
#pragma unroll
  for (int j = 0; j < 16; ++j) {
    vals[j] = __expf(vals[j] - m);
    sum += vals[j];
  }
#pragma unroll
  for (int off = 16; off > 0; off >>= 1) sum += __shfl_xor(sum, off, 32);

  const float inv = 1.0f / sum;
#pragma unroll
  for (int j = 0; j < 16; ++j)
    ls[wave * Cc + lane + 32 * j] = (__bf16)(vals[j] * inv);

  __syncthreads();

  // transposed write: thread owns 2 'd' columns, 8 consecutive 'c' -> b128
  const int b  = blockIdx.x >> 6;               // 64 blocks per batch
  const int c0 = (blockIdx.x & 63) * 8;
#pragma unroll
  for (int u = 0; u < 2; ++u) {
    const int d = threadIdx.x * 2 + u;
    v8bf pk;
#pragma unroll
    for (int r = 0; r < 8; ++r) pk[r] = ls[r * Cc + d];
    *(v8bf*)(aT + (size_t)b * Cc * Cc + (size_t)d * Cc + c0) = pk;
  }
}

// ---- Kernel 4: Out = x + gamma * (V[b] @ A[b]) ------------------------------
// aT is K-major [d][c]; V row-major [n][c] staged via async LDS.

__global__ __launch_bounds__(256) void out_gemm_kernel(const __bf16* __restrict__ vb,
                                                       const __bf16* __restrict__ aT,
                                                       const float* __restrict__ x,
                                                       const float* __restrict__ gamma,
                                                       float* __restrict__ out) {
  __shared__ __align__(16) __bf16 sA[2][16 * 32];

  const int tid  = threadIdx.x;
  const int wave = tid >> 5, lane = tid & 31;
  const int half = lane >> 4, l = lane & 15;

  const int mt = blockIdx.x;
  const int b  = blockIdx.y;
  const int col0 = wave * 64;

  const __bf16* Arow0 = vb + ((size_t)b * Npix + mt * 16) * Cc;
  const __bf16* Ab    = aT + (size_t)b * Cc * Cc;

  v8f z = {};
  v8f acc0 = z, acc1 = z, acc2 = z, acc3 = z;

  stage_tile(Arow0, Cc, sA[0], tid);
  int cur = 0;
  for (int k0 = 0; k0 < Cc; k0 += 32) {
    wait_async0();
    __syncthreads();
    if (k0 + 32 < Cc) stage_tile(Arow0 + k0 + 32, Cc, sA[cur ^ 1], tid);

    const v16bf a = load_a_frag_lds(sA[cur], l, half);
    const __bf16* bp = Ab + (size_t)(col0 + l) * Cc + k0 + half * 16;
    if (k0 + 64 < Cc) __builtin_prefetch(bp + 64, 0, 3);
    acc0 = wmma_bf16(a, load_b_frag_kmajor(bp + (size_t)0 * 16 * Cc), acc0);
    acc1 = wmma_bf16(a, load_b_frag_kmajor(bp + (size_t)1 * 16 * Cc), acc1);
    acc2 = wmma_bf16(a, load_b_frag_kmajor(bp + (size_t)2 * 16 * Cc), acc2);
    acc3 = wmma_bf16(a, load_b_frag_kmajor(bp + (size_t)3 * 16 * Cc), acc3);
    cur ^= 1;
  }

  const float g = gamma[0];
  v8f accs[4] = {acc0, acc1, acc2, acc3};
#pragma unroll
  for (int t = 0; t < 4; ++t) {
    const int col = col0 + t * 16 + l;
#pragma unroll
    for (int r = 0; r < 8; ++r) {
      const size_t idx = ((size_t)b * Npix + mt * 16 + half * 8 + r) * Cc + col;
      const float xr = __builtin_nontemporal_load(x + idx);
      __builtin_nontemporal_store(xr + g * accs[t][r], out + idx);
    }
  }
}

// ---- host launch -------------------------------------------------------------

extern "C" void kernel_launch(void* const* d_in, const int* in_sizes, int n_in,
                              void* d_out, int out_size, void* d_ws, size_t ws_size,
                              hipStream_t stream) {
  const float* x     = (const float*)d_in[0];
  const float* wq    = (const float*)d_in[1];
  const float* bq    = (const float*)d_in[2];
  const float* wk    = (const float*)d_in[3];
  const float* bk    = (const float*)d_in[4];
  const float* wv    = (const float*)d_in[5];
  const float* bv    = (const float*)d_in[6];
  const float* gamma = (const float*)d_in[7];
  float* out = (float*)d_out;

  // Workspace layout:
  //   xb       : B*N*C bf16   (x converted)
  //   qT, kT   : B*C*N bf16   (transposed, K-major)
  //   vb       : B*N*C bf16   (row-major)
  //   wbfT     : 3*C*C bf16   (K-major weights)
  //   s        : B*C*C f32
  //   aT       : B*C*C bf16   (K-major softmax output)
  char* ws = (char*)d_ws;
  const size_t big = (size_t)Bb * Npix * Cc * sizeof(__bf16);
  __bf16* xb   = (__bf16*)(ws);
  __bf16* qT   = (__bf16*)(ws + 1 * big);
  __bf16* kT   = (__bf16*)(ws + 2 * big);
  __bf16* vb   = (__bf16*)(ws + 3 * big);
  __bf16* wbfT = (__bf16*)(ws + 4 * big);
  char*   p    = ws + 4 * big + (size_t)3 * Cc * Cc * sizeof(__bf16);
  float*  s    = (float*)p;
  __bf16* aT   = (__bf16*)(p + (size_t)Bb * Cc * Cc * sizeof(float));

  const size_t xelems = (size_t)Bb * Npix * Cc;
  xconv_kernel<<<(unsigned)(xelems / (256 * 8)), 256, 0, stream>>>(x, xb);
  wconv_kernel<<<(3 * Cc * Cc) / 256, 256, 0, stream>>>(wq, wk, wv, wbfT);
  qkv_gemm_kernel<<<dim3(Npix / 16, Bb * 3), 256, 0, stream>>>(
      xb, wbfT, bq, bk, bv, qT, kT, vb);
  score_gemm_kernel<<<dim3(Cc / 16, Bb), 256, 0, stream>>>(qT, kT, s);
  softmax_kernel<<<(Bb * Cc) / 8, 256, 0, stream>>>(s, aT);
  out_gemm_kernel<<<dim3(Npix / 16, Bb), 256, 0, stream>>>(vb, aT, x, gamma, out);
}
